// MultiHeadAttention_13022340841511
// MI455X (gfx1250) — matrix operand order repaired and checked by reading, not compiled, not run on hardware
//
#include <hip/hip_runtime.h>
#include <hip/hip_bf16.h>
#include <stdint.h>
#include <stddef.h>

// ---------------------------------------------------------------------------
// MI455X (gfx1250) multi-head attention, bf16 WMMA path, wave32.
// ---------------------------------------------------------------------------

typedef __bf16 bf16_t;
typedef __attribute__((ext_vector_type(16))) __bf16 v16bf;
typedef __attribute__((ext_vector_type(8)))  float  v8f;

#define BATCH_ 2
#define SEQ_   2048
#define DIM_   1024
#define HEADS_ 16
#define DEPTH_ 64
#define ROWS_  (BATCH_ * SEQ_)   // 4096

union FragBits { v16bf v; unsigned u[8]; };

__device__ __forceinline__ bf16_t f2bf(float f) {
  union { unsigned u; float f; } a; a.f = f;
  unsigned r = (a.u + 0x7FFFu + ((a.u >> 16) & 1u)) >> 16;  // round-nearest-even
  union { unsigned short s; bf16_t b; } o; o.s = (unsigned short)r;
  return o.b;
}

// A fragment: 16x32 bf16, row-major source with leading dim `ld` (elements).
// ISA 7.12.2: lane m = L&15, half = L>>4; VGPR v holds K pair
// base = (v<4 ? 2v : 16+2(v-4)) + 8*half  (pair contiguous in memory).
__device__ __forceinline__ v16bf load_frag_a(const bf16_t* base, int ld) {
  int lane = threadIdx.x & 31;
  int m = lane & 15, half = lane >> 4;
  FragBits f;
#pragma unroll
  for (int v = 0; v < 8; ++v) {
    int kb = (v < 4 ? 2 * v : 16 + 2 * (v - 4)) + 8 * half;
    f.u[v] = *(const unsigned*)(base + (size_t)m * ld + kb);
  }
  return f.v;
}

// B fragment: 32x16 bf16. Source is "B transposed": row n (output col),
// contraction K contiguous with leading dim `ld`. ISA: lane n = L&15,
// half = L>>4; VGPR v holds K pair (v<4 ? 2v : 16+2(v-4)) + 8*half.
__device__ __forceinline__ v16bf load_frag_b(const bf16_t* base, int ld) {
  int lane = threadIdx.x & 31;
  int n = lane & 15, half = lane >> 4;
  FragBits f;
#pragma unroll
  for (int v = 0; v < 8; ++v) {
    int kb = (v < 4 ? 2 * v : 16 + 2 * (v - 4)) + 8 * half;
    f.u[v] = *(const unsigned*)(base + (size_t)n * ld + kb);
  }
  return f.v;
}

__device__ __forceinline__ v8f wmma_bf16(v16bf a, v16bf b, v8f c) {
  return __builtin_amdgcn_wmma_f32_16x16x32_bf16(false, a, false, b,
                                                 (short)0, c, false, false);
}

// 16-lane-group reductions (stays inside each half of the wave32).
// Only used once per wave after the streaming sweep.
__device__ __forceinline__ float rmax16(float v) {
  v = fmaxf(v, __shfl_xor(v, 1, 32));
  v = fmaxf(v, __shfl_xor(v, 2, 32));
  v = fmaxf(v, __shfl_xor(v, 4, 32));
  v = fmaxf(v, __shfl_xor(v, 8, 32));
  return v;
}
__device__ __forceinline__ float rsum16(float v) {
  v += __shfl_xor(v, 1, 32);
  v += __shfl_xor(v, 2, 32);
  v += __shfl_xor(v, 4, 32);
  v += __shfl_xor(v, 8, 32);
  return v;
}

// ---------------------------------------------------------------------------
// Elementwise conversions
// ---------------------------------------------------------------------------
__global__ __launch_bounds__(256) void f32_to_bf16_kernel(
    const float* __restrict__ x, bf16_t* __restrict__ y, size_t n) {
  size_t i = (size_t)blockIdx.x * 256 + threadIdx.x;
  if (i < n) y[i] = f2bf(x[i]);
}

// Wt[n*K + k] = bf16(W[k*N + n])   (K = N = 1024 here)
__global__ __launch_bounds__(256) void w_transpose_bf16_kernel(
    const float* __restrict__ W, bf16_t* __restrict__ Wt, int K, int N) {
  size_t i = (size_t)blockIdx.x * 256 + threadIdx.x;
  if (i >= (size_t)K * N) return;
  int k = (int)(i % K);
  int n = (int)(i / K);
  Wt[(size_t)n * K + k] = f2bf(W[(size_t)k * N + n]);
}

// Vt[bh][d][s] = Vp[(b*S+s)*D + h*64 + d]
__global__ __launch_bounds__(256) void v_transpose_kernel(
    const bf16_t* __restrict__ Vp, bf16_t* __restrict__ Vt) {
  size_t i = (size_t)blockIdx.x * 256 + threadIdx.x;
  if (i >= (size_t)BATCH_ * HEADS_ * DEPTH_ * SEQ_) return;
  int s  = (int)(i & (SEQ_ - 1));
  size_t j = i >> 11;             // /SEQ_
  int d  = (int)(j & (DEPTH_ - 1));
  int bh = (int)(j >> 6);
  int b = bh / HEADS_, h = bh % HEADS_;
  Vt[i] = Vp[((size_t)(b * SEQ_ + s)) * DIM_ + h * DEPTH_ + d];
}

// ---------------------------------------------------------------------------
// WMMA GEMM: C[M,N] = (A[M,K] @ B[K,N] + bias) * outScale,
// B supplied transposed (N x K). Double-buffered fragment loads.
// Block = 256 thr = 8 waves; block tile 32x256, wave tile 16x64.
// ---------------------------------------------------------------------------
template <bool OUT_BF16>
__global__ __launch_bounds__(256) void gemm_wmma_kernel(
    const bf16_t* __restrict__ A, const bf16_t* __restrict__ Bt,
    const float* __restrict__ bias, void* __restrict__ Cout,
    int M, int N, int K, float outScale) {
  int wave = threadIdx.x >> 5;
  int lane = threadIdx.x & 31;
  int wr = wave >> 2, wc = wave & 3;
  int tilesN = N >> 8;
  int bm = blockIdx.x / tilesN, bn = blockIdx.x % tilesN;
  int row0 = bm * 32 + wr * 16;
  int col0 = bn * 256 + wc * 64;

  const bf16_t* Arow = A + (size_t)row0 * K;

  v8f acc[4] = {v8f{}, v8f{}, v8f{}, v8f{}};
  v16bf a = load_frag_a(Arow, K);
  v16bf b[4];
#pragma unroll
  for (int t = 0; t < 4; ++t)
    b[t] = load_frag_b(Bt + (size_t)(col0 + t * 16) * K, K);

  for (int k = 0; k < K; k += 32) {
    int kn = (k + 32 < K) ? k + 32 : k;       // clamp: last iter reloads (dead)
    v16bf an = load_frag_a(Arow + kn, K);
    v16bf bn_[4];
#pragma unroll
    for (int t = 0; t < 4; ++t)
      bn_[t] = load_frag_b(Bt + (size_t)(col0 + t * 16) * K + kn, K);
#pragma unroll
    for (int t = 0; t < 4; ++t)
      acc[t] = wmma_bf16(a, b[t], acc[t]);
    a = an;
#pragma unroll
    for (int t = 0; t < 4; ++t) b[t] = bn_[t];
  }

  int n0 = lane & 15, half = lane >> 4;
#pragma unroll
  for (int t = 0; t < 4; ++t) {
    int col = col0 + t * 16 + n0;
    float bv = bias[col];
#pragma unroll
    for (int r = 0; r < 8; ++r) {
      int row = row0 + r + 8 * half;
      float val = (acc[t][r] + bv) * outScale;
      if (OUT_BF16) ((bf16_t*)Cout)[(size_t)row * N + col] = f2bf(val);
      else          ((float*)Cout)[(size_t)row * N + col]  = val;
    }
  }
}

// ---------------------------------------------------------------------------
// Fused attention: per wave, one 16-row q-tile of one (b,h).
// Q is pre-scaled by 1/sqrt(depth) in its projection, so WMMA output = logits.
// Sweep 1: QK^T (WMMA) -> PER-LANE online max/sum-exp (no cross-lane traffic),
//          single 16-lane combine at the end. Double-buffered K fragments.
// Sweep 2: recompute logits, write normalized attn (f32, write-once),
//          relayout P via wave-private LDS, ctx += P@V (WMMA).
// Block = 8 waves = 128 q rows; grid = B*H * (S/128) = 512.
// ---------------------------------------------------------------------------
__global__ __launch_bounds__(256) void attention_kernel(
    const bf16_t* __restrict__ Q, const bf16_t* __restrict__ Kb,
    const bf16_t* __restrict__ Vt, float* __restrict__ attnOut,
    bf16_t* __restrict__ ctxOut) {
  __shared__ alignas(16) bf16_t pbuf[8][16 * 32];

  int wave = threadIdx.x >> 5;
  int lane = threadIdx.x & 31;
  int bh   = blockIdx.x >> 4;            // /(S/128)=16
  int qblk = blockIdx.x & 15;
  int b = bh / HEADS_, h = bh % HEADS_;
  int q0 = qblk * 128 + wave * 16;
  int n = lane & 15, half = lane >> 4;

  const bf16_t* Qp  = Q  + ((size_t)(b * SEQ_ + q0)) * DIM_ + h * DEPTH_;
  const bf16_t* Kp  = Kb + ((size_t)b * SEQ_) * DIM_ + h * DEPTH_;
  const bf16_t* Vtp = Vt + (size_t)bh * DEPTH_ * SEQ_;

  v16bf aq0 = load_frag_a(Qp, DIM_);
  v16bf aq1 = load_frag_a(Qp + 32, DIM_);

  // ---- sweep 1: per-lane online softmax statistics (zero shuffles inside) ----
  float m[8], l[8];
#pragma unroll
  for (int r = 0; r < 8; ++r) { m[r] = -1e30f; l[r] = 0.f; }

  const int JT = SEQ_ / 16;
  v16bf bk0 = load_frag_b(Kp, DIM_);
  v16bf bk1 = load_frag_b(Kp + 32, DIM_);
  for (int j = 0; j < JT; ++j) {
    if (j + 4 < JT)
      __builtin_prefetch(Kp + (size_t)((j + 4) * 16 + n) * DIM_, 0, 0);
    int jn = (j + 1 < JT) ? j + 1 : j;        // clamp: last iter reloads (dead)
    v16bf nb0 = load_frag_b(Kp + (size_t)(jn * 16) * DIM_, DIM_);
    v16bf nb1 = load_frag_b(Kp + (size_t)(jn * 16) * DIM_ + 32, DIM_);
    v8f c = {};
    c = wmma_bf16(aq0, bk0, c);
    c = wmma_bf16(aq1, bk1, c);
#pragma unroll
    for (int r = 0; r < 8; ++r) {
      float t  = c[r];
      float mo = m[r];
      float e  = __expf(-fabsf(t - mo));    // exp(min-max), one transcendental
      bool  gt = t > mo;
      m[r] = gt ? t : mo;
      l[r] = gt ? (l[r] * e + 1.f) : (l[r] + e);
    }
    bk0 = nb0;
    bk1 = nb1;
  }
  // one-time 16-lane combine: row max + adjusted row sum
  float mrow[8], rl[8];
#pragma unroll
  for (int r = 0; r < 8; ++r) {
    float M = rmax16(m[r]);
    float L = rsum16(l[r] * __expf(m[r] - M));
    mrow[r] = M;
    rl[r] = 1.f / L;
  }

  // ---- sweep 2: write attn + ctx accumulation (double-buffered K frags) ----
  v8f accv[4] = {v8f{}, v8f{}, v8f{}, v8f{}};
  bf16_t* pb = pbuf[wave];
  float* attnBase = attnOut + ((size_t)bh * SEQ_ + q0) * SEQ_;

  const int J2 = SEQ_ / 32;
  v16bf ck0 = load_frag_b(Kp, DIM_);
  v16bf ck1 = load_frag_b(Kp + 32, DIM_);
  for (int j2 = 0; j2 < J2; ++j2) {
    if (j2 + 2 < J2) {
      __builtin_prefetch(Kp + (size_t)((j2 + 2) * 32 + n) * DIM_, 0, 0);
      __builtin_prefetch(Vtp + (size_t)n * SEQ_ + (j2 + 2) * 32, 0, 0);
    }
#pragma unroll
    for (int sub = 0; sub < 2; ++sub) {
      int j = j2 * 2 + sub;
      int jn = (j + 1 < 2 * J2) ? j + 1 : j;  // clamp: last iter reloads (dead)
      v16bf nk0 = load_frag_b(Kp + (size_t)(jn * 16) * DIM_, DIM_);
      v16bf nk1 = load_frag_b(Kp + (size_t)(jn * 16) * DIM_ + 32, DIM_);
      v8f c = {};
      c = wmma_bf16(aq0, ck0, c);
      c = wmma_bf16(aq1, ck1, c);
#pragma unroll
      for (int r = 0; r < 8; ++r) {
        float p = __expf(c[r] - mrow[r]) * rl[r];
        int row = r + 8 * half;
        attnBase[(size_t)row * SEQ_ + j * 16 + n] = p;       // write-once attn
        pb[row * 32 + sub * 16 + n] = f2bf(p);               // C->A relayout
      }
      ck0 = nk0;
      ck1 = nk1;
    }
    // LDS buffer is wave-private; DS ops are in-order within a wave, so a
    // compiler scheduling fence suffices (no s_barrier needed).
    __builtin_amdgcn_wave_barrier();
    v16bf ap = load_frag_a(pb, 32);
    int s0 = j2 * 32;
#pragma unroll
    for (int t = 0; t < 4; ++t) {
      v16bf bv = load_frag_b(Vtp + (size_t)(t * 16) * SEQ_ + s0, SEQ_);
      accv[t] = wmma_bf16(ap, bv, accv[t]);
    }
    __builtin_amdgcn_wave_barrier();
  }

#pragma unroll
  for (int t = 0; t < 4; ++t)
#pragma unroll
    for (int r = 0; r < 8; ++r) {
      int row = q0 + r + 8 * half;
      int col = h * DEPTH_ + t * 16 + n;
      ctxOut[(size_t)(b * SEQ_ + row) * DIM_ + col] = f2bf(accv[t][r]);
    }
}

// ---------------------------------------------------------------------------
// LayerNorm over last dim (N=1024), one block per row.
// ---------------------------------------------------------------------------
__global__ __launch_bounds__(256) void layernorm_kernel(
    const float* __restrict__ x, const float* __restrict__ gamma,
    const float* __restrict__ beta, float* __restrict__ y) {
  const int N = DIM_;
  int row = blockIdx.x;
  const float* xr = x + (size_t)row * N;
  float s = 0.f, ss = 0.f;
  for (int i = threadIdx.x; i < N; i += 256) { float v = xr[i]; s += v; ss += v * v; }
  __shared__ float r1[256], r2[256];
  r1[threadIdx.x] = s; r2[threadIdx.x] = ss;
  __syncthreads();
  for (int off = 128; off > 0; off >>= 1) {
    if ((int)threadIdx.x < off) {
      r1[threadIdx.x] += r1[threadIdx.x + off];
      r2[threadIdx.x] += r2[threadIdx.x + off];
    }
    __syncthreads();
  }
  float mu  = r1[0] * (1.f / N);
  float var = r2[0] * (1.f / N) - mu * mu;
  float inv = rsqrtf(var + 1e-3f);
  for (int i = threadIdx.x; i < N; i += 256)
    y[(size_t)row * N + i] = (xr[i] - mu) * inv * gamma[i] + beta[i];
}

// ---------------------------------------------------------------------------
// Host-side orchestration (graph-capture safe: launches only).
// ---------------------------------------------------------------------------
extern "C" void kernel_launch(void* const* d_in, const int* in_sizes, int n_in,
                              void* d_out, int out_size, void* d_ws, size_t ws_size,
                              hipStream_t stream) {
  (void)in_sizes; (void)n_in; (void)out_size; (void)ws_size;

  const float* query = (const float*)d_in[0];
  const float* key   = (const float*)d_in[1];
  const float* value = (const float*)d_in[2];
  const float* Wq    = (const float*)d_in[3];
  const float* bq    = (const float*)d_in[4];
  const float* Wk    = (const float*)d_in[5];
  const float* bk    = (const float*)d_in[6];
  const float* Wv    = (const float*)d_in[7];
  const float* bv    = (const float*)d_in[8];
  const float* Wo    = (const float*)d_in[9];
  const float* bo    = (const float*)d_in[10];
  const float* gamma = (const float*)d_in[11];
  const float* beta  = (const float*)d_in[12];

  const size_t MB = 1u << 20;
  char* ws = (char*)d_ws;
  // phase-1 regions [0,24MB) reused later for the pre-LN projection (f32, 16MB)
  bf16_t* qin  = (bf16_t*)(ws + 0 * MB);
  bf16_t* kin  = (bf16_t*)(ws + 8 * MB);
  bf16_t* vin  = (bf16_t*)(ws + 16 * MB);
  float*  proj = (float*)(ws + 0 * MB);     // reuse after q/k/v consumed
  bf16_t* Wqt  = (bf16_t*)(ws + 24 * MB);
  bf16_t* Wkt  = (bf16_t*)(ws + 26 * MB);
  bf16_t* Wvt  = (bf16_t*)(ws + 28 * MB);
  bf16_t* Wot  = (bf16_t*)(ws + 30 * MB);
  bf16_t* Qp   = (bf16_t*)(ws + 32 * MB);
  bf16_t* Kp   = (bf16_t*)(ws + 40 * MB);
  bf16_t* Vp   = (bf16_t*)(ws + 48 * MB);
  bf16_t* Vt   = (bf16_t*)(ws + 56 * MB);
  bf16_t* ctx  = (bf16_t*)(ws + 64 * MB);   // 72 MB total

  float* outY    = (float*)d_out;
  float* attnOut = (float*)d_out + (size_t)ROWS_ * DIM_;  // after `out`

  const size_t nAct = (size_t)ROWS_ * DIM_;   // 4,194,304
  const int cvtBlocks = (int)((nAct + 255) / 256);

  f32_to_bf16_kernel<<<cvtBlocks, 256, 0, stream>>>(query, qin, nAct);
  f32_to_bf16_kernel<<<cvtBlocks, 256, 0, stream>>>(key,   kin, nAct);
  f32_to_bf16_kernel<<<cvtBlocks, 256, 0, stream>>>(value, vin, nAct);

  const size_t nW = (size_t)DIM_ * DIM_;
  const int wBlocks = (int)((nW + 255) / 256);
  w_transpose_bf16_kernel<<<wBlocks, 256, 0, stream>>>(Wq, Wqt, DIM_, DIM_);
  w_transpose_bf16_kernel<<<wBlocks, 256, 0, stream>>>(Wk, Wkt, DIM_, DIM_);
  w_transpose_bf16_kernel<<<wBlocks, 256, 0, stream>>>(Wv, Wvt, DIM_, DIM_);
  w_transpose_bf16_kernel<<<wBlocks, 256, 0, stream>>>(Wo, Wot, DIM_, DIM_);

  const int gemmGrid = (ROWS_ / 32) * (DIM_ / 256);  // 512
  // Q projection folds in 1/sqrt(depth) = 0.125 (exact power of two in bf16).
  gemm_wmma_kernel<true><<<gemmGrid, 256, 0, stream>>>(qin, Wqt, bq, Qp, ROWS_, DIM_, DIM_, 0.125f);
  gemm_wmma_kernel<true><<<gemmGrid, 256, 0, stream>>>(kin, Wkt, bk, Kp, ROWS_, DIM_, DIM_, 1.0f);
  gemm_wmma_kernel<true><<<gemmGrid, 256, 0, stream>>>(vin, Wvt, bv, Vp, ROWS_, DIM_, DIM_, 1.0f);

  v_transpose_kernel<<<cvtBlocks, 256, 0, stream>>>(Vp, Vt);

  const int attnGrid = BATCH_ * HEADS_ * (SEQ_ / 128);  // 512
  attention_kernel<<<attnGrid, 256, 0, stream>>>(Qp, Kp, Vt, attnOut, ctx);

  gemm_wmma_kernel<false><<<gemmGrid, 256, 0, stream>>>(ctx, Wot, bo, proj, ROWS_, DIM_, DIM_, 1.0f);

  layernorm_kernel<<<ROWS_, 256, 0, stream>>>(proj, gamma, beta, outY);
}
